// BWN_13554916786219
// MI455X (gfx1250) — compile-verified, matrix operand order checked by reading, not run on hardware
//
#include <hip/hip_runtime.h>

// ---------------------------------------------------------------------------
// BWN MLP on MI455X (gfx1250):
//   y = sigmoid(relu(relu(x @ (a1*sgnW1)^T + b1) @ (a2*sgnW2)^T + b2) @ (a3*sgnW3)^T + b3)
// Phase 1: alphas via deterministic reduction; operands pre-converted to bf16
//          (weights binarized to +-1 bf16 exactly).
// Phase 2: bf16 WMMA GEMMs; global->LDS staging via global_load_async_to_lds_b128
//          (ASYNCcnt), double-buffered; alpha/bias/activation fused epilogue.
// ---------------------------------------------------------------------------

#define BM 128
#define BN 128
#define BK 32
#define KDIM 4096         // K is 4096 for all three layers (compile-time)
#define LDT 40            // padded LDS row stride in bf16 elements (32 + 8 pad)
#define NTHREADS 128      // 4 waves of 32

typedef __attribute__((ext_vector_type(16))) __bf16 v16bf;
typedef __attribute__((ext_vector_type(8)))  __bf16 v8bf;
typedef __attribute__((ext_vector_type(4)))  __bf16 v4bf;
typedef __attribute__((ext_vector_type(8)))  float  v8f;

// ---------------- async staging: global -> LDS, 128 rows x 32 cols bf16 -----
// Each thread issues 4 x 16B async copies; tracked by ASYNCcnt, no VGPR data.

__device__ __forceinline__ void stage_async_tile(const __bf16* __restrict__ G, long rowBase,
                                                 int kBase, __bf16* __restrict__ s, int t) {
#pragma unroll
  for (int it = 0; it < 4; ++it) {
    const int row = it * 32 + (t >> 2);
    const int col = (t & 3) * 8;
    const unsigned long long gp =
        (unsigned long long)(const void*)(G + (rowBase + row) * (long)KDIM + kBase + col);
    const unsigned lds = (unsigned)(unsigned long long)(const void*)(s + row * LDT + col);
    asm volatile("global_load_async_to_lds_b128 %0, %1, off"
                 :: "v"(lds), "v"(gp) : "memory");
  }
}

__device__ __forceinline__ void wait_async_all() {
  asm volatile("s_wait_asynccnt 0x0" ::: "memory");
}

// ---------------- LDS -> WMMA fragment (16x32 bf16, wave32 layout) ----------
// Lanes 0-15: row = lane, K[0..7] + K[16..23]; lanes 16-31: same rows, K[8..15] + K[24..31].

__device__ __forceinline__ v16bf ldfrag(const __bf16* __restrict__ p0, int lane) {
  const int r  = lane & 15;
  const int hi = (lane >> 4) & 1;
  const __bf16* p = p0 + r * LDT + hi * 8;
  v8bf lo = *(const v8bf*)(p);
  v8bf hh = *(const v8bf*)(p + 16);
  return __builtin_shufflevector(lo, hh, 0, 1, 2, 3, 4, 5, 6, 7, 8, 9, 10, 11, 12, 13, 14, 15);
}

// ---------------- GEMM: out = act(alpha * (A @ Wb^T) + bias) ----------------
// A: (M,4096) bf16 row-major; Wb: (N,4096) bf16 (+-1) row-major; ACT 0=relu, 1=sigmoid.

template <int ACT, typename OT>
__global__ __launch_bounds__(NTHREADS) void gemm_bwn(
    const __bf16* __restrict__ A, const __bf16* __restrict__ Wb,
    const float* __restrict__ bias, const float* __restrict__ alphaPtr,
    OT* __restrict__ out, int N) {
  __shared__ __align__(16) __bf16 sA[2][BM * LDT];
  __shared__ __align__(16) __bf16 sB[2][BN * LDT];

  const int t    = threadIdx.x;
  const int lane = t & 31;
  const int wid  = t >> 5;
  const int wmo  = (wid >> 1) * 64;  // wave M offset in macro tile
  const int wno  = (wid & 1) * 64;   // wave N offset in macro tile
  const long mBlock = (long)blockIdx.y * BM;
  const long nBlock = (long)blockIdx.x * BN;

  v8f acc[4][4] = {};

  // Preload tile 0 (async -> LDS), then enter steady-state double buffer.
  stage_async_tile(A,  mBlock, 0, sA[0], t);
  stage_async_tile(Wb, nBlock, 0, sB[0], t);
  wait_async_all();
  __syncthreads();

  const int NT = KDIM / BK;
  for (int kt = 0; kt < NT; ++kt) {
    const int cur = kt & 1;

    // Kick off next tile's async copies first so they overlap the WMMA work.
    if (kt + 1 < NT) {
      stage_async_tile(A,  mBlock, (kt + 1) * BK, sA[cur ^ 1], t);
      stage_async_tile(Wb, nBlock, (kt + 1) * BK, sB[cur ^ 1], t);
    }

    const __bf16* aB = &sA[cur][wmo * LDT];
    const __bf16* bB = &sB[cur][wno * LDT];

    v16bf bf[4];
#pragma unroll
    for (int ni = 0; ni < 4; ++ni) bf[ni] = ldfrag(bB + ni * 16 * LDT, lane);

#pragma unroll
    for (int mi = 0; mi < 4; ++mi) {
      v16bf af = ldfrag(aB + mi * 16 * LDT, lane);
#pragma unroll
      for (int ni = 0; ni < 4; ++ni)
        acc[mi][ni] = __builtin_amdgcn_wmma_f32_16x16x32_bf16(
            false, af, false, bf[ni], (short)0, acc[mi][ni], false, false);
    }

    wait_async_all();   // next tile fully landed in LDS
    __syncthreads();    // visible to all waves before it becomes 'cur'
  }

  // Epilogue. C/D layout: lanes 0-15: (M = r, N = lane); lanes 16-31: (M = r+8, N = lane-16).
  const float alpha = alphaPtr[0];
  const int nl = lane & 15;
  const int hi = (lane >> 4) & 1;
  const long mW = mBlock + wmo;
  const long nW = nBlock + wno;

#pragma unroll
  for (int ni = 0; ni < 4; ++ni) {
    const long n  = nW + ni * 16 + nl;
    const float bv = bias[n];
#pragma unroll
    for (int mi = 0; mi < 4; ++mi) {
#pragma unroll
      for (int r = 0; r < 8; ++r) {
        const long m = mW + mi * 16 + r + hi * 8;
        const float z = alpha * acc[mi][ni][r] + bv;
        float y;
        if (ACT == 0) {
          y = z > 0.f ? z : 0.f;
        } else {
          y = 1.0f / (1.0f + __expf(-z));
        }
        out[m * (long)N + n] = (OT)y;
      }
    }
  }
}

// ---------------- operand pre-conversion (bandwidth-trivial) ----------------

__global__ __launch_bounds__(256) void bwn_cvt_bf16(const float* __restrict__ in,
                                                    __bf16* __restrict__ out, int n4) {
  const int i = blockIdx.x * blockDim.x + threadIdx.x;
  if (i < n4) {
    const float4 v = ((const float4*)in)[i];
    v4bf o;
    o[0] = (__bf16)v.x; o[1] = (__bf16)v.y; o[2] = (__bf16)v.z; o[3] = (__bf16)v.w;
    *(v4bf*)(out + (long)i * 4) = o;
  }
}

__global__ __launch_bounds__(256) void bwn_binarize(const float* __restrict__ in,
                                                    __bf16* __restrict__ out, int n4) {
  const int i = blockIdx.x * blockDim.x + threadIdx.x;
  if (i < n4) {
    const float4 v = ((const float4*)in)[i];
    v4bf o;
    o[0] = v.x >= 0.f ? (__bf16)1.0f : (__bf16)(-1.0f);
    o[1] = v.y >= 0.f ? (__bf16)1.0f : (__bf16)(-1.0f);
    o[2] = v.z >= 0.f ? (__bf16)1.0f : (__bf16)(-1.0f);
    o[3] = v.w >= 0.f ? (__bf16)1.0f : (__bf16)(-1.0f);
    *(v4bf*)(out + (long)i * 4) = o;
  }
}

// ---------------- alpha = mean(|W|): deterministic two-stage reduction ------

__global__ __launch_bounds__(256) void bwn_abssum_partial(const float* __restrict__ w, int n,
                                                          float* __restrict__ part) {
  float s = 0.f;
  const int stride = gridDim.x * blockDim.x;
  for (int i = blockIdx.x * blockDim.x + threadIdx.x; i < n; i += stride) s += fabsf(w[i]);
#pragma unroll
  for (int o = 16; o > 0; o >>= 1) s += __shfl_xor(s, o);
  __shared__ float red[8];
  if ((threadIdx.x & 31) == 0) red[threadIdx.x >> 5] = s;
  __syncthreads();
  if (threadIdx.x == 0) {
    float tt = 0.f;
#pragma unroll
    for (int i = 0; i < 8; ++i) tt += red[i];
    part[blockIdx.x] = tt;
  }
}

__global__ __launch_bounds__(256) void bwn_absmean_final(const float* __restrict__ part, int n,
                                                         float scale, float* __restrict__ alpha) {
  float s = 0.f;
  for (int i = threadIdx.x; i < n; i += blockDim.x) s += part[i];
#pragma unroll
  for (int o = 16; o > 0; o >>= 1) s += __shfl_xor(s, o);
  __shared__ float red[8];
  if ((threadIdx.x & 31) == 0) red[threadIdx.x >> 5] = s;
  __syncthreads();
  if (threadIdx.x == 0) {
    float tt = 0.f;
#pragma unroll
    for (int i = 0; i < 8; ++i) tt += red[i];
    alpha[0] = tt * scale;
  }
}

// ---------------------------------------------------------------------------

extern "C" void kernel_launch(void* const* d_in, const int* in_sizes, int n_in,
                              void* d_out, int out_size, void* d_ws, size_t ws_size,
                              hipStream_t stream) {
  constexpr long BATCH = 8192, INS = 4096, HID = 4096, OUTS = 1024;

  const float* x  = (const float*)d_in[0];
  const float* w1 = (const float*)d_in[1];
  const float* b1 = (const float*)d_in[2];
  const float* w2 = (const float*)d_in[3];
  const float* b2 = (const float*)d_in[4];
  const float* w3 = (const float*)d_in[5];
  const float* b3 = (const float*)d_in[6];
  float* out = (float*)d_out;

  // Workspace layout (bytes)
  char* ws = (char*)d_ws;
  float*  alphas = (float*)ws;           // 3 floats
  float*  part   = (float*)(ws + 256);   // 1024 floats, reused per layer
  size_t off = 8192;
  __bf16* xb  = (__bf16*)(ws + off); off += (size_t)BATCH * INS * 2;   // 64 MiB
  __bf16* h1  = (__bf16*)(ws + off); off += (size_t)BATCH * HID * 2;   // 64 MiB
  __bf16* h2  = (__bf16*)(ws + off); off += (size_t)BATCH * HID * 2;   // 64 MiB
  __bf16* w1b = (__bf16*)(ws + off); off += (size_t)HID * INS * 2;     // 32 MiB
  __bf16* w2b = (__bf16*)(ws + off); off += (size_t)HID * HID * 2;     // 32 MiB
  __bf16* w3b = (__bf16*)(ws + off);                                    // 8 MiB

  // alphas (sequential kernels on one stream => deterministic)
  bwn_abssum_partial<<<1024, 256, 0, stream>>>(w1, (int)(HID * INS), part);
  bwn_absmean_final<<<1, 256, 0, stream>>>(part, 1024, 1.0f / (float)(HID * INS), alphas + 0);
  bwn_abssum_partial<<<1024, 256, 0, stream>>>(w2, (int)(HID * HID), part);
  bwn_absmean_final<<<1, 256, 0, stream>>>(part, 1024, 1.0f / (float)(HID * HID), alphas + 1);
  bwn_abssum_partial<<<1024, 256, 0, stream>>>(w3, (int)(OUTS * HID), part);
  bwn_absmean_final<<<1, 256, 0, stream>>>(part, 1024, 1.0f / (float)(OUTS * HID), alphas + 2);

  // one-time bf16 conversion / binarization
  {
    const int nx = (int)(BATCH * INS / 4);
    bwn_cvt_bf16<<<(nx + 255) / 256, 256, 0, stream>>>(x, xb, nx);
    const int n1 = (int)(HID * INS / 4);
    bwn_binarize<<<(n1 + 255) / 256, 256, 0, stream>>>(w1, w1b, n1);
    const int n2 = (int)(HID * HID / 4);
    bwn_binarize<<<(n2 + 255) / 256, 256, 0, stream>>>(w2, w2b, n2);
    const int n3 = (int)(OUTS * HID / 4);
    bwn_binarize<<<(n3 + 255) / 256, 256, 0, stream>>>(w3, w3b, n3);
  }

  // layer 1: h1 = relu(a1 * (xb @ w1b^T) + b1)
  gemm_bwn<0, __bf16><<<dim3(HID / BN, BATCH / BM), NTHREADS, 0, stream>>>(
      xb, w1b, b1, alphas + 0, h1, (int)HID);
  // layer 2: h2 = relu(a2 * (h1 @ w2b^T) + b2)
  gemm_bwn<0, __bf16><<<dim3(HID / BN, BATCH / BM), NTHREADS, 0, stream>>>(
      h1, w2b, b2, alphas + 1, h2, (int)HID);
  // layer 3: out = sigmoid(a3 * (h2 @ w3b^T) + b3)
  gemm_bwn<1, float><<<dim3(OUTS / BN, BATCH / BM), NTHREADS, 0, stream>>>(
      h2, w3b, b3, alphas + 2, out, (int)OUTS);
}
